// AdvancedTemporalGraphNetwork_14723147891194
// MI455X (gfx1250) — compile-verified, matrix operand order checked by reading, not run on hardware
//
#include <hip/hip_runtime.h>
#include <cstdint>

#define NN 50000
#define FF 256
#define HH 128
#define EE 800000
#define BB 64
#define TT 50

typedef __attribute__((ext_vector_type(16))) __bf16 v16bf;
typedef __attribute__((ext_vector_type(8)))  float  v8f;

union FragB { uint32_t u[8]; v16bf v; };

__device__ __forceinline__ unsigned short f2bf(float x) {
  union { float f; uint32_t u; } t; t.f = x;
  return (unsigned short)((t.u + 0x7FFFu + ((t.u >> 16) & 1u)) >> 16);
}
__device__ __forceinline__ uint32_t pack2bf(float lo, float hi) {
  return (uint32_t)f2bf(lo) | ((uint32_t)f2bf(hi) << 16);
}
// 16-bit A/B fragment K-pattern (CDNA5 ISA 7.12.2): VGPR v, lane l -> K pair base
__device__ __forceinline__ int kpat(int v, int lane) {
  return ((v >= 4) ? 16 : 0) + ((v & 3) << 1) + ((lane >> 4) << 3);
}
__device__ __forceinline__ float sigm(float x) { return 1.0f / (1.0f + __expf(-x)); }
// scheduling fence: forbids hoisting/merging memory ops across it (register
// values are unaffected) -> bounds live fragment count, prevents spills
__device__ __forceinline__ void sched_fence() { asm volatile("" ::: "memory"); }

// ---------------------------------------------------------------------------
// Generic WMMA GEMM: D[M,N] = act(A[M,K]_f32 @ W[N,K]^T_bf16 + bias + Cadd)
// block = 256 thr (8 waves), tile = 128(M) x 128(N), K-step 32, bf16 WMMA.
// ---------------------------------------------------------------------------
#define ASTR 34  // LDS row stride in shorts (68B, 4B-aligned, conflict-breaking)

template<bool RELU, bool CADD, bool BIAS>
__global__ __launch_bounds__(256)
void gemm_wmma(const float* __restrict__ A, int lda,
               const unsigned short* __restrict__ W,  // [N,K] bf16 row-major
               const float* __restrict__ bias,
               const float* __restrict__ Cadd, int ldc,
               float* __restrict__ D, int ldd,
               int M, int N, int K)
{
  __shared__ unsigned short As[128 * ASTR];
  __shared__ unsigned short Ws[128 * ASTR];
  const int tid  = threadIdx.x;
  const int lane = tid & 31;
  const int w    = tid >> 5;
  const int mbase = blockIdx.x * 128;
  const int nbase = blockIdx.y * 128;

  v8f acc[8];
#pragma unroll
  for (int j = 0; j < 8; ++j)
#pragma unroll
    for (int r = 0; r < 8; ++r) acc[j][r] = 0.0f;

  const int ksteps = K >> 5;
  for (int ks = 0; ks < ksteps; ++ks) {
    // stage A tile (128 x 32 f32 -> bf16 LDS), coalesced float4
#pragma unroll
    for (int i = 0; i < 4; ++i) {
      int idx = i * 256 + tid;            // 1024 float4
      int row = idx >> 3;
      int k4  = (idx & 7) << 2;
      int gr  = mbase + row;
      float4 val = make_float4(0.f, 0.f, 0.f, 0.f);
      if (gr < M) {
        val = *(const float4*)(A + (size_t)gr * lda + (ks << 5) + k4);
        if (ks + 1 < ksteps)
          __builtin_prefetch(A + (size_t)gr * lda + ((ks + 1) << 5) + k4, 0, 1);
      }
      uint32_t* p = (uint32_t*)&As[row * ASTR + k4];
      p[0] = pack2bf(val.x, val.y);
      p[1] = pack2bf(val.z, val.w);
    }
    // stage W tile (128 x 32 bf16 -> LDS), coalesced 32-bit
#pragma unroll
    for (int i = 0; i < 8; ++i) {
      int idx = i * 256 + tid;            // 2048 uint
      int row = idx >> 4;
      int k2  = idx & 15;
      int n   = nbase + row;
      uint32_t u = 0u;
      if (n < N) u = *(const uint32_t*)(W + (size_t)n * K + (ks << 5) + (k2 << 1));
      *(uint32_t*)&Ws[row * ASTR + (k2 << 1)] = u;
    }
    __syncthreads();

    FragB af;
#pragma unroll
    for (int v = 0; v < 8; ++v) {
      int r  = (w << 4) + (lane & 15);
      int kk = kpat(v, lane);
      af.u[v] = *(const uint32_t*)&As[r * ASTR + kk];
    }
#pragma unroll
    for (int j = 0; j < 8; ++j) {
      FragB bf;
#pragma unroll
      for (int v = 0; v < 8; ++v) {
        int r  = (j << 4) + (lane & 15);
        int kk = kpat(v, lane);
        bf.u[v] = *(const uint32_t*)&Ws[r * ASTR + kk];
      }
      acc[j] = __builtin_amdgcn_wmma_f32_16x16x32_bf16(
          false, af.v, false, bf.v, (short)0, acc[j], false, false);
    }
    __syncthreads();
  }

  // epilogue: C/D layout = VGPR r: lanes0-15 -> M=r, lanes16-31 -> M=r+8; N=lane&15
#pragma unroll
  for (int j = 0; j < 8; ++j) {
    int col = nbase + (j << 4) + (lane & 15);
#pragma unroll
    for (int r = 0; r < 8; ++r) {
      int row = mbase + (w << 4) + r + ((lane >> 4) << 3);
      if (row < M && col < N) {
        float val = acc[j][r];
        if (BIAS) val += bias[col];
        if (CADD) val += Cadd[(size_t)row * ldc + col];
        if (RELU) val = fmaxf(val, 0.0f);
        D[(size_t)row * ldd + col] = val;
      }
    }
  }
}

// ---------------------------------------------------------------------------
// Graph mean-aggregation kernels
// ---------------------------------------------------------------------------
__global__ void fill_k(float* __restrict__ p, float v, int n4) {
  int i = blockIdx.x * 256 + threadIdx.x;
  if (i < n4) { float4 t = make_float4(v, v, v, v); ((float4*)p)[i] = t; }
}

__global__ void edge_count_k(const int* __restrict__ dst, float* __restrict__ cnt, int E) {
  int e = blockIdx.x * 256 + threadIdx.x;
  if (e < E) atomicAdd(&cnt[dst[e]], 1.0f);
}

// one wave covers one edge's 128 floats (4 per lane) -> coalesced 512B rows
__global__ void edge_scatter_k(const float* __restrict__ xin,
                               const int* __restrict__ src, const int* __restrict__ dst,
                               float* __restrict__ agg, int E) {
  int gid = blockIdx.x * 256 + threadIdx.x;
  if (gid >= E * 32) return;
  int e  = gid >> 5;
  int f4 = (gid & 31) << 2;
  int s = src[e], d = dst[e];
  float4 v = *(const float4*)(xin + (size_t)s * HH + f4);
  float* p = agg + (size_t)d * HH + f4;
  atomicAdd(p + 0, v.x); atomicAdd(p + 1, v.y);
  atomicAdd(p + 2, v.z); atomicAdd(p + 3, v.w);
}

__global__ void normalize_k(float* __restrict__ agg, const float* __restrict__ cnt, int n4) {
  int i = blockIdx.x * 256 + threadIdx.x;
  if (i >= n4) return;
  int node = i >> 5;                       // 32 float4 per node (128 feats)
  float c = fmaxf(cnt[node], 1.0f);
  float4 v = ((const float4*)agg)[i];
  v.x /= c; v.y /= c; v.z /= c; v.w /= c;
  ((float4*)agg)[i] = v;
}

__global__ void cvt_bf16_k(const float* __restrict__ s, unsigned short* __restrict__ d, int n) {
  int i = blockIdx.x * 256 + threadIdx.x;
  if (i < n) d[i] = f2bf(s[i]);
}

// ---------------------------------------------------------------------------
// Persistent LSTM layer: one workgroup, 16 waves (512 thr), T sequential steps.
// gates[64,512] = x_t@Wih^T + h@Whh^T + bih + bhh  (bf16 WMMA, K=[x|h]=256)
// x pre-converted to bf16 (no pack in hot loop); h (bf16) & c (f32) in LDS.
// Wave w: rows (w&3)*16..+15, cols (w>>2)*128..+127 -> 8 WMMA tiles.
// sched_fence() between tiles stops the scheduler from batching all B
// fragments (which previously spilled to scratch).
// ---------------------------------------------------------------------------
#define HSTR 132
#define LSTM_THREADS 512
__global__ __launch_bounds__(LSTM_THREADS)
void lstm_layer_k(const unsigned short* __restrict__ xbf,  // [B*T,128] bf16, rows b*T+t
                  const unsigned short* __restrict__ Wih,  // [512,128] bf16
                  const unsigned short* __restrict__ Whh,  // [512,128] bf16
                  const float* __restrict__ bih, const float* __restrict__ bhh,
                  float* __restrict__ y,                   // [B*T,128] f32
                  float* __restrict__ gates)               // [64,512] f32 scratch
{
  __shared__ unsigned short Hs[64 * HSTR]; // h, bf16
  __shared__ float Cs[64 * 128];           // c, f32
  const int tid  = threadIdx.x;
  const int lane = tid & 31;
  const int w    = tid >> 5;               // 0..15
  const int rstrip = (w & 3) << 4;         // 4 row strips of 16
  const int cgrp   = (w >> 2) << 7;        // 4 col groups of 128

  for (int i = tid; i < 64 * 128; i += LSTM_THREADS) {
    int b = i >> 7, hc = i & 127;
    Hs[b * HSTR + hc] = 0;
    Cs[i] = 0.0f;
  }
  __syncthreads();

  for (int t = 0; t < TT; ++t) {
    v8f acc[8];
#pragma unroll
    for (int j = 0; j < 8; ++j)
#pragma unroll
      for (int r = 0; r < 8; ++r) acc[j][r] = 0.0f;

#pragma unroll
    for (int ks = 0; ks < 8; ++ks) {
      FragB af;
      int brow = rstrip + (lane & 15);
      if (ks < 4) { // A = x_t slice, bf16 from global
#pragma unroll
        for (int v = 0; v < 8; ++v) {
          int kk = (ks << 5) + kpat(v, lane);
          af.u[v] = *(const uint32_t*)(xbf + (size_t)(brow * TT + t) * 128 + kk);
        }
      } else {      // A = h, bf16 from LDS
#pragma unroll
        for (int v = 0; v < 8; ++v) {
          int kk = ((ks - 4) << 5) + kpat(v, lane);
          af.u[v] = *(const uint32_t*)&Hs[brow * HSTR + kk];
        }
      }
      sched_fence();
      const unsigned short* Wsel = (ks < 4) ? Wih : Whh;
      const int kb = (ks & 3) << 5;
#pragma unroll
      for (int j = 0; j < 8; ++j) {
        int n = cgrp + (j << 4) + (lane & 15);
        FragB bf;
#pragma unroll
        for (int v = 0; v < 8; ++v) {
          int kk = kb + kpat(v, lane);
          bf.u[v] = *(const uint32_t*)(Wsel + (size_t)n * 128 + kk);
        }
        acc[j] = __builtin_amdgcn_wmma_f32_16x16x32_bf16(
            false, af.v, false, bf.v, (short)0, acc[j], false, false);
        sched_fence();
      }
    }
    // write gates (+ both biases) to global scratch
#pragma unroll
    for (int j = 0; j < 8; ++j) {
      int col = cgrp + (j << 4) + (lane & 15);
#pragma unroll
      for (int r = 0; r < 8; ++r) {
        int row = rstrip + r + ((lane >> 4) << 3);
        gates[row * 512 + col] = acc[j][r] + bih[col] + bhh[col];
      }
    }
    __threadfence();
    __syncthreads();
    // cell update
    for (int i = tid; i < 64 * 128; i += LSTM_THREADS) {
      int b = i >> 7, hc = i & 127;
      float gi = gates[b * 512 + hc];
      float gf = gates[b * 512 + 128 + hc];
      float gg = gates[b * 512 + 256 + hc];
      float go = gates[b * 512 + 384 + hc];
      float c = sigm(gf) * Cs[i] + sigm(gi) * tanhf(gg);
      float h = sigm(go) * tanhf(c);
      Cs[i] = c;
      Hs[b * HSTR + hc] = f2bf(h);
      y[(size_t)(b * TT + t) * 128 + hc] = h;
    }
    __syncthreads();
  }
}

// ---------------------------------------------------------------------------
// MHA: grid (B, NH), block 64. dh=64, T=50. K/V staged in LDS.
// ---------------------------------------------------------------------------
__global__ __launch_bounds__(64)
void mha_k(const float* __restrict__ qkv, float* __restrict__ o) {
  __shared__ float Ks[TT][64];
  __shared__ float Vs[TT][64];
  const int b = blockIdx.x, h = blockIdx.y;
  const int tid = threadIdx.x;
  for (int r = 0; r < TT; ++r) {
    Ks[r][tid] = qkv[(size_t)(b * TT + r) * 384 + 128 + h * 64 + tid];
    Vs[r][tid] = qkv[(size_t)(b * TT + r) * 384 + 256 + h * 64 + tid];
  }
  __syncthreads();
  if (tid < TT) {
    float q[64];
#pragma unroll
    for (int d = 0; d < 64; ++d)
      q[d] = qkv[(size_t)(b * TT + tid) * 384 + h * 64 + d];
    float sc[TT];
    float mx = -1e30f;
    for (int r = 0; r < TT; ++r) {
      float dot = 0.0f;
#pragma unroll
      for (int d = 0; d < 64; ++d) dot += q[d] * Ks[r][d];
      dot *= 0.125f; // 1/sqrt(64)
      sc[r] = dot;
      mx = fmaxf(mx, dot);
    }
    float sum = 0.0f;
    for (int r = 0; r < TT; ++r) { sc[r] = __expf(sc[r] - mx); sum += sc[r]; }
    float inv = 1.0f / sum;
    float out[64];
#pragma unroll
    for (int d = 0; d < 64; ++d) out[d] = 0.0f;
    for (int r = 0; r < TT; ++r) {
      float a = sc[r] * inv;
#pragma unroll
      for (int d = 0; d < 64; ++d) out[d] += a * Vs[r][d];
    }
#pragma unroll
    for (int d = 0; d < 64; ++d)
      o[(size_t)(b * TT + tid) * 128 + h * 64 + d] = out[d];
  }
}

// ---------------------------------------------------------------------------
extern "C" void kernel_launch(void* const* d_in, const int* in_sizes, int n_in,
                              void* d_out, int out_size, void* d_ws, size_t ws_size,
                              hipStream_t stream) {
  (void)in_sizes; (void)n_in; (void)out_size; (void)ws_size;
  const float* x      = (const float*)d_in[0];
  const int*   ei     = (const int*)d_in[1];
  const int*   srcI   = ei;
  const int*   dstI   = ei + EE;
  const float* tseq   = (const float*)d_in[2];
  const float* W_red  = (const float*)d_in[3];
  const float* b_red  = (const float*)d_in[4];
  const float* s1Wl   = (const float*)d_in[5];
  const float* s1bl   = (const float*)d_in[6];
  const float* s1Wr   = (const float*)d_in[7];
  const float* s2Wl   = (const float*)d_in[8];
  const float* s2bl   = (const float*)d_in[9];
  const float* s2Wr   = (const float*)d_in[10];
  const float* Wih0   = (const float*)d_in[11];
  const float* Whh0   = (const float*)d_in[12];
  const float* bih0   = (const float*)d_in[13];
  const float* bhh0   = (const float*)d_in[14];
  const float* Wih1   = (const float*)d_in[15];
  const float* Whh1   = (const float*)d_in[16];
  const float* bih1   = (const float*)d_in[17];
  const float* bhh1   = (const float*)d_in[18];
  const float* Win    = (const float*)d_in[19];
  const float* binA   = (const float*)d_in[20];
  const float* Wout   = (const float*)d_in[21];
  const float* bout   = (const float*)d_in[22];
  const float* hW1    = (const float*)d_in[23];
  const float* hb1    = (const float*)d_in[24];
  const float* hW2    = (const float*)d_in[25];
  const float* hb2    = (const float*)d_in[26];
  float* out = (float*)d_out;          // [0:64)=pred, [64: ) = spatial (50000x128)

  // workspace layout (f32 units)
  float* f = (float*)d_ws;
  float* reduced = f;               f += (size_t)NN * HH;
  float* agg     = f;               f += (size_t)NN * HH;
  float* tmp     = f;               f += (size_t)NN * HH;
  float* sbuf    = f;               f += (size_t)NN * HH;
  float* cnt     = f;               f += NN;
  float* ts_red  = f;               f += (size_t)BB * TT * HH;
  float* y0      = f;               f += (size_t)BB * TT * HH;
  float* y1      = f;               f += (size_t)BB * TT * HH;
  float* qkv     = f;               f += (size_t)BB * TT * 384;
  float* attno   = f;               f += (size_t)BB * TT * HH;
  float* attnout = f;               f += (size_t)BB * TT * HH;
  float* gates   = f;               f += 64 * 512;
  float* headt   = f;               f += 64 * 64;
  unsigned short* bw = (unsigned short*)f;
  unsigned short* W_red_b = bw;     bw += 128 * 256;
  unsigned short* s1Wl_b  = bw;     bw += 128 * 128;
  unsigned short* s1Wr_b  = bw;     bw += 128 * 128;
  unsigned short* s2Wl_b  = bw;     bw += 128 * 128;
  unsigned short* s2Wr_b  = bw;     bw += 128 * 128;
  unsigned short* Wih0_b  = bw;     bw += 512 * 128;
  unsigned short* Whh0_b  = bw;     bw += 512 * 128;
  unsigned short* Wih1_b  = bw;     bw += 512 * 128;
  unsigned short* Whh1_b  = bw;     bw += 512 * 128;
  unsigned short* Win_b   = bw;     bw += 384 * 128;
  unsigned short* Wout_b  = bw;     bw += 128 * 128;
  unsigned short* hW1_b   = bw;     bw += 64 * 128;
  unsigned short* hW2_b   = bw;     bw += 64;
  unsigned short* ts_bf   = bw;     bw += BB * TT * HH;  // lstm0 input, bf16
  unsigned short* y0_bf   = bw;     bw += BB * TT * HH;  // lstm1 input, bf16

  auto cdiv = [](int a, int b) { return (a + b - 1) / b; };
  auto cvt = [&](const float* s, unsigned short* d, int n) {
    cvt_bf16_k<<<cdiv(n, 256), 256, 0, stream>>>(s, d, n);
  };
  cvt(W_red, W_red_b, 128 * 256);
  cvt(s1Wl, s1Wl_b, 128 * 128); cvt(s1Wr, s1Wr_b, 128 * 128);
  cvt(s2Wl, s2Wl_b, 128 * 128); cvt(s2Wr, s2Wr_b, 128 * 128);
  cvt(Wih0, Wih0_b, 512 * 128); cvt(Whh0, Whh0_b, 512 * 128);
  cvt(Wih1, Wih1_b, 512 * 128); cvt(Whh1, Whh1_b, 512 * 128);
  cvt(Win, Win_b, 384 * 128);   cvt(Wout, Wout_b, 128 * 128);
  cvt(hW1, hW1_b, 64 * 128);    cvt(hW2, hW2_b, 64);

  const int mtN = cdiv(NN, 128);      // 391
  const int mtS = cdiv(BB * TT, 128); // 25

  // reduced = x @ W_red^T + b_red
  gemm_wmma<false, false, true><<<dim3(mtN, 1), 256, 0, stream>>>(
      x, FF, W_red_b, b_red, nullptr, 0, reduced, HH, NN, HH, FF);

  // degree counts (dst-only, reused for both SAGE layers)
  fill_k<<<cdiv(NN / 4, 256), 256, 0, stream>>>(cnt, 0.0f, NN / 4);
  edge_count_k<<<cdiv(EE, 256), 256, 0, stream>>>(dstI, cnt, EE);

  // SAGE 1
  fill_k<<<cdiv(NN * HH / 4, 256), 256, 0, stream>>>(agg, 0.0f, NN * HH / 4);
  edge_scatter_k<<<cdiv(EE * 32, 256), 256, 0, stream>>>(reduced, srcI, dstI, agg, EE);
  normalize_k<<<cdiv(NN * HH / 4, 256), 256, 0, stream>>>(agg, cnt, NN * HH / 4);
  gemm_wmma<false, false, true><<<dim3(mtN, 1), 256, 0, stream>>>(
      agg, HH, s1Wl_b, s1bl, nullptr, 0, tmp, HH, NN, HH, HH);
  gemm_wmma<true, true, false><<<dim3(mtN, 1), 256, 0, stream>>>(
      reduced, HH, s1Wr_b, nullptr, tmp, HH, sbuf, HH, NN, HH, HH);

  // SAGE 2 -> spatial output at out+64
  fill_k<<<cdiv(NN * HH / 4, 256), 256, 0, stream>>>(agg, 0.0f, NN * HH / 4);
  edge_scatter_k<<<cdiv(EE * 32, 256), 256, 0, stream>>>(sbuf, srcI, dstI, agg, EE);
  normalize_k<<<cdiv(NN * HH / 4, 256), 256, 0, stream>>>(agg, cnt, NN * HH / 4);
  gemm_wmma<false, false, true><<<dim3(mtN, 1), 256, 0, stream>>>(
      agg, HH, s2Wl_b, s2bl, nullptr, 0, tmp, HH, NN, HH, HH);
  gemm_wmma<true, true, false><<<dim3(mtN, 1), 256, 0, stream>>>(
      sbuf, HH, s2Wr_b, nullptr, tmp, HH, out + 64, HH, NN, HH, HH);

  // temporal branch: ts_red = time_sequence @ W_red^T + b_red
  gemm_wmma<false, false, true><<<dim3(mtS, 1), 256, 0, stream>>>(
      tseq, FF, W_red_b, b_red, nullptr, 0, ts_red, HH, BB * TT, HH, FF);

  // two LSTM layers (persistent single-workgroup kernels, bf16 inputs)
  cvt(ts_red, ts_bf, BB * TT * HH);
  lstm_layer_k<<<1, LSTM_THREADS, 0, stream>>>(ts_bf, Wih0_b, Whh0_b, bih0, bhh0, y0, gates);
  cvt(y0, y0_bf, BB * TT * HH);
  lstm_layer_k<<<1, LSTM_THREADS, 0, stream>>>(y0_bf, Wih1_b, Whh1_b, bih1, bhh1, y1, gates);

  // attention
  gemm_wmma<false, false, true><<<dim3(mtS, 3), 256, 0, stream>>>(
      y1, HH, Win_b, binA, nullptr, 0, qkv, 384, BB * TT, 384, HH);
  mha_k<<<dim3(BB, 2), 64, 0, stream>>>(qkv, attno);
  gemm_wmma<false, false, true><<<dim3(mtS, 1), 256, 0, stream>>>(
      attno, HH, Wout_b, bout, nullptr, 0, attnout, HH, BB * TT, HH, HH);

  // head on last timestep (strided A)
  gemm_wmma<true, false, true><<<dim3(1, 1), 256, 0, stream>>>(
      attnout + (TT - 1) * HH, TT * HH, hW1_b, hb1, nullptr, 0, headt, 64, BB, 64, HH);
  gemm_wmma<false, false, true><<<dim3(1, 1), 256, 0, stream>>>(
      headt, 64, hW2_b, hb2, nullptr, 0, out, 1, BB, 1, 64);
}
